// SparseVoxelNet_27487790695149
// MI455X (gfx1250) — matrix operand order repaired and checked by reading, not compiled, not run on hardware
//
#include <hip/hip_runtime.h>
#include <hip/hip_bf16.h>

// ---------- types ----------
typedef __attribute__((ext_vector_type(16))) __bf16       bf16x16;
typedef __attribute__((ext_vector_type(8)))  float        floatx8;
typedef __attribute__((ext_vector_type(4)))  unsigned int uint32x4;

union V16 {                 // one WMMA A/B operand per lane (16 bf16 = 32B)
    bf16x16      v;
    uint32x4     q[2];
    unsigned short h[16];
};

__device__ __forceinline__ unsigned short f2bf(float x) {
    unsigned u = __float_as_uint(x);
    unsigned r = u + 0x7FFFu + ((u >> 16) & 1u);   // round-to-nearest-even
    return (unsigned short)(r >> 16);
}

// ---------- weight pre-pack into WMMA-B register layout ----------
// packed[(k*KS+ks)*NT+nt][lane][e] = bf16(W[k][ks*32 + 16*(lane>>4) + e][nt*16 + (lane&15)])
__global__ void pack_w_kernel(const float* __restrict__ W, unsigned short* __restrict__ pw,
                              int NOFF, int CIN, int COUT)
{
    int KS = CIN / 32, NT = COUT / 16;
    int total = NOFF * KS * NT * 512;
    for (int i = blockIdx.x * blockDim.x + threadIdx.x; i < total;
         i += gridDim.x * blockDim.x) {
        int e    = i & 15;
        int lane = (i >> 4) & 31;
        int t    = i >> 9;            // (k*KS+ks)*NT + nt
        int nt   = t % NT;
        int ks   = (t / NT) % KS;
        int k    = t / (NT * KS);
        int ci   = ks * 32 + 16 * (lane >> 4) + e;
        int co   = nt * 16 + (lane & 15);
        pw[i] = f2bf(W[((size_t)k * CIN + ci) * COUT + co]);
    }
}

// ---------- expand (parent idx, offset id) into 8-offset neighbor table ----------
__global__ void build_upnbr_kernel(const int* __restrict__ pidx, const int* __restrict__ poff,
                                   int* __restrict__ nbr, int N)
{
    int i = blockIdx.x * blockDim.x + threadIdx.x;
    if (i >= N) return;
    int pi = pidx[i], po = poff[i];
#pragma unroll
    for (int k = 0; k < 8; ++k) nbr[(size_t)k * N + i] = (po == k) ? pi : -1;
}

// ---------- layer 1: Cin=2 is too small for WMMA; plain VALU ----------
__global__ void conv1_kernel(const float* __restrict__ feats, const int* __restrict__ nbr,
                             const float* __restrict__ W1, float* __restrict__ out, int N0)
{
    int i = blockIdx.x * blockDim.x + threadIdx.x;
    int row = i >> 5, co = i & 31;
    if (row >= N0) return;
    float s = 0.f;
#pragma unroll
    for (int k = 0; k < 27; ++k) {
        int idx = nbr[(size_t)k * N0 + row];
        if (idx >= 0) {
            float f0 = feats[(size_t)idx * 2 + 0];
            float f1 = feats[(size_t)idx * 2 + 1];
            s = fmaf(f0, W1[(k * 2 + 0) * 32 + co], s);
            s = fmaf(f1, W1[(k * 2 + 1) * 32 + co], s);
        }
    }
    out[(size_t)row * 32 + co] = s;
}

// ---------- WMMA implicit-GEMM sparse conv, M-blocked, LDS-staged B ----------
// One wave computes MB x (16-row) x COUT output tiles.
// A (16x32 bf16) layout: lane = 16*h + r holds row r, channels [16h,16h+16) (matches packed B).
// B slab for the current kernel offset k is staged in LDS by the whole workgroup
// (KS*NT*1KB <= 16KB), then each wave reads tiles with ds_load_b128.
// C/D layout: lane l -> col l&15, VGPR v -> row 8*(l>>4)+v.
template <int CIN, int COUT, int NOFF, int MB>
__global__ __launch_bounds__(256) void sconv_wmma_kernel(
    const unsigned short* __restrict__ fh,   // [Nin, CIN] bf16
    const int*            __restrict__ nbr,  // [NOFF, Nout]
    const unsigned short* __restrict__ pw,   // packed weights
    float*                __restrict__ out,  // [Nout, COUT] fp32
    int Nout)
{
    constexpr int KS = CIN / 32;
    constexpr int NT = COUT / 16;
    constexpr int SLAB16 = KS * NT * 64;     // 16B chunks per k-slab

    __shared__ unsigned short sB[KS * NT * 512];   // <= 16 KB

    int tid  = threadIdx.x;
    int lane = tid & 31;
    int wave = blockIdx.x * (blockDim.x >> 5) + (tid >> 5);
    int row0 = wave * 16 * MB;
    // NOTE: no early return — all waves must reach the barriers below.

    int r = lane & 15;
    int h = lane >> 4;

    bool rowok[MB];
    int  rowc[MB];
#pragma unroll
    for (int mb = 0; mb < MB; ++mb) {
        int row = row0 + mb * 16 + r;
        rowok[mb] = row < Nout;
        rowc[mb]  = rowok[mb] ? row : (Nout - 1);   // safe clamp for tail loads
    }

    const uint32x4 zq = {0u, 0u, 0u, 0u};
    floatx8 acc[MB][NT];
#pragma unroll
    for (int mb = 0; mb < MB; ++mb)
#pragma unroll
        for (int nt = 0; nt < NT; ++nt) acc[mb][nt] = (floatx8)0.0f;

    for (int k = 0; k < NOFF; ++k) {
        // --- stage this offset's packed B slab into LDS (whole workgroup) ---
        __syncthreads();                                   // protect previous slab reads
        const uint32x4* pwkq = (const uint32x4*)(pw + (size_t)k * KS * NT * 512);
        uint32x4* sBq = (uint32x4*)sB;
        for (int i = tid; i < SLAB16; i += 256) sBq[i] = pwkq[i];
        __syncthreads();

        // --- gather A slabs for MB row tiles ---
        V16 a[MB][KS];
#pragma unroll
        for (int mb = 0; mb < MB; ++mb) {
            int idx = nbr[(size_t)k * Nout + rowc[mb]];
            if (idx >= 0 && rowok[mb]) {
                const unsigned short* fr = fh + (size_t)idx * CIN;
#pragma unroll
                for (int ks = 0; ks < KS; ++ks) {
                    int c0 = ks * 32 + 16 * h;
                    a[mb][ks].q[0] = *(const uint32x4*)(fr + c0);
                    a[mb][ks].q[1] = *(const uint32x4*)(fr + c0 + 8);
                }
            } else {
#pragma unroll
                for (int ks = 0; ks < KS; ++ks) { a[mb][ks].q[0] = zq; a[mb][ks].q[1] = zq; }
            }
        }

        // --- B from LDS, MB WMMAs per tile ---
#pragma unroll
        for (int ks = 0; ks < KS; ++ks) {
#pragma unroll
            for (int nt = 0; nt < NT; ++nt) {
                V16 b;
                const unsigned short* pb = sB + ((ks * NT + nt) * 32 + lane) * 16;
                b.q[0] = *(const uint32x4*)(pb);
                b.q[1] = *(const uint32x4*)(pb + 8);
#pragma unroll
                for (int mb = 0; mb < MB; ++mb) {
                    acc[mb][nt] = __builtin_amdgcn_wmma_f32_16x16x32_bf16(
                        false, a[mb][ks].v, false, b.v, (short)0, acc[mb][nt], false, false);
                }
            }
        }
    }

#pragma unroll
    for (int mb = 0; mb < MB; ++mb) {
#pragma unroll
        for (int nt = 0; nt < NT; ++nt) {
            int co = nt * 16 + r;
#pragma unroll
            for (int v = 0; v < 8; ++v) {
                int orow = row0 + mb * 16 + 8 * h + v;
                if (orow < Nout) out[(size_t)orow * COUT + co] = acc[mb][nt][v];
            }
        }
    }
}

// ---------- BatchNorm stats: per-channel sum / sumsq via float atomics ----------
__global__ __launch_bounds__(256) void bn_stats_kernel(const float* __restrict__ X,
                                                       float* __restrict__ stats,
                                                       int N, int C)
{
    int tid = blockIdx.x * blockDim.x + threadIdx.x;
    int T = gridDim.x * blockDim.x;          // multiple of C by construction
    int c = tid % C;
    int r0 = tid / C;
    int rs = T / C;
    float s = 0.f, s2 = 0.f;
    for (int r = r0; r < N; r += rs) {
        float v = X[(size_t)r * C + c];
        s += v; s2 += v * v;
    }
    atomicAdd(&stats[c], s);
    atomicAdd(&stats[256 + c], s2);
}

// ---------- BN normalize + ReLU + bf16 (and fp32) write ----------
__global__ void bn_apply_kernel(const float* __restrict__ X, const float* __restrict__ stats,
                                const float* __restrict__ g, const float* __restrict__ b,
                                unsigned short* __restrict__ Yh, float* __restrict__ Yf,
                                int N, int C)
{
    size_t i = (size_t)blockIdx.x * blockDim.x + threadIdx.x;
    size_t total = (size_t)N * C;
    if (i >= total) return;
    int c = (int)(i % C);
    float inv_n = 1.0f / (float)N;
    float mu  = stats[c] * inv_n;
    float var = stats[256 + c] * inv_n - mu * mu;
    float y = (X[i] - mu) * rsqrtf(var + 1e-5f) * g[c] + b[c];
    y = fmaxf(y, 0.f);
    Yh[i] = f2bf(y);
    if (Yf) Yf[i] = y;
}

// ---------- final 1x1 conv head: [N0,32] @ [32,2] ----------
__global__ void head_kernel(const float* __restrict__ d1, const float* __restrict__ Wout,
                            float* __restrict__ out, int N0)
{
    int i = blockIdx.x * blockDim.x + threadIdx.x;
    int row = i >> 1, j = i & 1;
    if (row >= N0) return;
    float s = 0.f;
#pragma unroll
    for (int c = 0; c < 32; ++c) s = fmaf(d1[(size_t)row * 32 + c], Wout[c * 2 + j], s);
    out[(size_t)row * 2 + j] = s;
}

// =====================================================================
extern "C" void kernel_launch(void* const* d_in, const int* in_sizes, int n_in,
                              void* d_out, int out_size, void* d_ws, size_t ws_size,
                              hipStream_t stream)
{
    const float* feats = (const float*)d_in[0];
    const float* W1    = (const float*)d_in[1];
    const float* W2    = (const float*)d_in[2];
    const float* W3    = (const float*)d_in[3];
    const float* W4    = (const float*)d_in[4];
    const float* W5    = (const float*)d_in[5];
    const float* Wout  = (const float*)d_in[6];
    const float* g1 = (const float*)d_in[7],  *b1 = (const float*)d_in[8];
    const float* g2 = (const float*)d_in[9],  *b2 = (const float*)d_in[10];
    const float* g3 = (const float*)d_in[11], *b3 = (const float*)d_in[12];
    const float* g4 = (const float*)d_in[13], *b4 = (const float*)d_in[14];
    const float* g5 = (const float*)d_in[15], *b5 = (const float*)d_in[16];
    const int* nbr0 = (const int*)d_in[17];
    const int* nbr1 = (const int*)d_in[18];
    const int* nbr2 = (const int*)d_in[19];
    const int* up1_idx = (const int*)d_in[20];
    const int* up1_off = (const int*)d_in[21];
    const int* up0_idx = (const int*)d_in[22];
    const int* up0_off = (const int*)d_in[23];

    const int N0 = in_sizes[0] / 2;
    const int N1 = in_sizes[20];
    const int N2 = in_sizes[19] / 27;

    // ---- workspace layout (256B aligned slabs) ----
    char* ws = (char*)d_ws;
    size_t off = 0;
    auto alloc = [&](size_t bytes) -> char* {
        char* p = ws + off;
        off += (bytes + 255) & ~(size_t)255;
        return p;
    };
    size_t xmax = (size_t)N0 * 32;
    if ((size_t)N1 * 64  > xmax) xmax = (size_t)N1 * 64;
    if ((size_t)N2 * 128 > xmax) xmax = (size_t)N2 * 128;

    float*          X    = (float*)alloc(xmax * sizeof(float));        // reused conv output
    unsigned short* H1   = (unsigned short*)alloc((size_t)N0 * 32 * 2);
    unsigned short* H2   = (unsigned short*)alloc((size_t)N1 * 64 * 2);
    unsigned short* H3   = (unsigned short*)alloc((size_t)N2 * 128 * 2);
    unsigned short* H4   = (unsigned short*)alloc((size_t)N1 * 64 * 2);
    float*          D1F  = (float*)alloc((size_t)N0 * 32 * sizeof(float));
    unsigned short* PW2  = (unsigned short*)alloc((size_t)27 * 1 * 4 * 512 * 2);
    unsigned short* PW3  = (unsigned short*)alloc((size_t)27 * 2 * 8 * 512 * 2);
    unsigned short* PW4  = (unsigned short*)alloc((size_t)8  * 4 * 4 * 512 * 2);
    unsigned short* PW5  = (unsigned short*)alloc((size_t)8  * 2 * 2 * 512 * 2);
    int*            NU1  = (int*)alloc((size_t)8 * N1 * sizeof(int));
    int*            NU0  = (int*)alloc((size_t)8 * N0 * sizeof(int));
    float*          ST   = (float*)alloc(512 * sizeof(float));         // [sum(256) | sumsq(256)]

    const int TB = 256;
    auto blocks_for = [](long long n, int tb) { return (int)((n + tb - 1) / tb); };
    auto conv_blocks = [](int nrows, int mb) {
        long long waves = ((long long)nrows + 16LL * mb - 1) / (16LL * mb);
        return (int)((waves + 7) / 8);                    // 8 waves / block
    };
    const int statBlocks = 128;  // 32768 threads; multiple of every C used

    // ---- prep: pack weights, expand up-conv maps ----
    pack_w_kernel<<<blocks_for(27LL*1*4*512, TB), TB, 0, stream>>>(W2, PW2, 27, 32, 64);
    pack_w_kernel<<<blocks_for(27LL*2*8*512, TB), TB, 0, stream>>>(W3, PW3, 27, 64, 128);
    pack_w_kernel<<<blocks_for(8LL*4*4*512,  TB), TB, 0, stream>>>(W4, PW4, 8, 128, 64);
    pack_w_kernel<<<blocks_for(8LL*2*2*512,  TB), TB, 0, stream>>>(W5, PW5, 8, 64, 32);
    build_upnbr_kernel<<<blocks_for(N1, TB), TB, 0, stream>>>(up1_idx, up1_off, NU1, N1);
    build_upnbr_kernel<<<blocks_for(N0, TB), TB, 0, stream>>>(up0_idx, up0_off, NU0, N0);

    // ---- e1 = ReLU(BN(conv1)) ----
    conv1_kernel<<<blocks_for((long long)N0 * 32, TB), TB, 0, stream>>>(feats, nbr0, W1, X, N0);
    hipMemsetAsync(ST, 0, 512 * sizeof(float), stream);
    bn_stats_kernel<<<statBlocks, TB, 0, stream>>>(X, ST, N0, 32);
    bn_apply_kernel<<<blocks_for((long long)N0 * 32, TB), TB, 0, stream>>>(X, ST, g1, b1, H1, nullptr, N0, 32);

    // ---- e2: WMMA sconv 32->64 over 27 offsets (MB=4) ----
    sconv_wmma_kernel<32, 64, 27, 4><<<conv_blocks(N1, 4), TB, 0, stream>>>(H1, nbr1, PW2, X, N1);
    hipMemsetAsync(ST, 0, 512 * sizeof(float), stream);
    bn_stats_kernel<<<statBlocks, TB, 0, stream>>>(X, ST, N1, 64);
    bn_apply_kernel<<<blocks_for((long long)N1 * 64, TB), TB, 0, stream>>>(X, ST, g2, b2, H2, nullptr, N1, 64);

    // ---- e3: WMMA sconv 64->128 over 27 offsets (MB=2) ----
    sconv_wmma_kernel<64, 128, 27, 2><<<conv_blocks(N2, 2), TB, 0, stream>>>(H2, nbr2, PW3, X, N2);
    hipMemsetAsync(ST, 0, 512 * sizeof(float), stream);
    bn_stats_kernel<<<statBlocks, TB, 0, stream>>>(X, ST, N2, 128);
    bn_apply_kernel<<<blocks_for((long long)N2 * 128, TB), TB, 0, stream>>>(X, ST, g3, b3, H3, nullptr, N2, 128);

    // ---- d2: WMMA transposed conv 128->64 (8 offsets, one active per row, MB=2) ----
    sconv_wmma_kernel<128, 64, 8, 2><<<conv_blocks(N1, 2), TB, 0, stream>>>(H3, NU1, PW4, X, N1);
    hipMemsetAsync(ST, 0, 512 * sizeof(float), stream);
    bn_stats_kernel<<<statBlocks, TB, 0, stream>>>(X, ST, N1, 64);
    bn_apply_kernel<<<blocks_for((long long)N1 * 64, TB), TB, 0, stream>>>(X, ST, g4, b4, H4, nullptr, N1, 64);

    // ---- d1: WMMA transposed conv 64->32 (MB=4) ----
    sconv_wmma_kernel<64, 32, 8, 4><<<conv_blocks(N0, 4), TB, 0, stream>>>(H4, NU0, PW5, X, N0);
    hipMemsetAsync(ST, 0, 512 * sizeof(float), stream);
    bn_stats_kernel<<<statBlocks, TB, 0, stream>>>(X, ST, N0, 32);
    bn_apply_kernel<<<blocks_for((long long)N0 * 32, TB), TB, 0, stream>>>(X, ST, g5, b5, H1 /*scratch*/, D1F, N0, 32);

    // ---- head: [N0,32] @ [32,2] ----
    head_kernel<<<blocks_for((long long)N0 * 2, TB), TB, 0, stream>>>(D1F, Wout, (float*)d_out, N0);
}
